// GQAAttention_39204461477977
// MI455X (gfx1250) — matrix operand order repaired and checked by reading, not compiled, not run on hardware
//
#include <hip/hip_runtime.h>
#include <hip/hip_bf16.h>
#include <math.h>

typedef __attribute__((ext_vector_type(16))) __bf16 v16bf;
typedef __attribute__((ext_vector_type(8)))  __bf16 v8bf;
typedef __attribute__((ext_vector_type(4)))  __bf16 v4bf;
typedef __attribute__((ext_vector_type(8)))  float  v8f;

#define HIDDEN 2048
#define SEQ    2048
#define BATCH  2
#define NH     32
#define NKV    8
#define HD     64

#define LDS_ROW 48   // 96-byte padded row stride (mult of 32B for aligned v16bf reads)

#define WMMA_BF16(a, b, c) \
  __builtin_amdgcn_wmma_f32_16x16x32_bf16(false, (a), false, (b), (short)0, (c), false, false)

static __device__ __forceinline__ v16bf load_a_frag(const __bf16* p0, const __bf16* p1) {
  v8bf lo = *(const v8bf*)p0;
  v8bf hi = *(const v8bf*)p1;
  v16bf r;
#pragma unroll
  for (int i = 0; i < 8; ++i) { r[i] = lo[i]; r[i + 8] = hi[i]; }
  return r;
}

static __device__ __forceinline__ v16bf load_b_frag(const __bf16* p) {
  return *(const v16bf*)p;
}

// ---- CDNA5 async global->LDS copies (ASYNCcnt-tracked, bypass VGPRs) ----
// Same inst_offset is added to both LDS and global addresses (ISA 08 §4.4).
static __device__ __forceinline__ void async_copy64B(unsigned lds, const __bf16* g) {
  asm volatile("global_load_async_to_lds_b128 %0, %1, off\n\t"
               "global_load_async_to_lds_b128 %0, %1, off offset:16\n\t"
               "global_load_async_to_lds_b128 %0, %1, off offset:32\n\t"
               "global_load_async_to_lds_b128 %0, %1, off offset:48"
               :: "v"(lds), "v"(g) : "memory");
}
static __device__ __forceinline__ void async_copy32B(unsigned lds, const __bf16* g) {
  asm volatile("global_load_async_to_lds_b128 %0, %1, off\n\t"
               "global_load_async_to_lds_b128 %0, %1, off offset:16"
               :: "v"(lds), "v"(g) : "memory");
}
static __device__ __forceinline__ void async_wait6() {
  asm volatile("s_wait_asynccnt 0x6" ::: "memory");
}
static __device__ __forceinline__ void async_wait0() {
  asm volatile("s_wait_asynccnt 0x0" ::: "memory");
}
static __device__ __forceinline__ unsigned lds_off(const void* p) {
  return (unsigned)(unsigned long long)(uintptr_t)p;  // generic LDS ptr: low 32 bits = LDS offset
}

// ---------------- fp32 -> bf16 conversion (vectorized) ----------------
__global__ void cvt_bf16_kernel(const float* __restrict__ in, __bf16* __restrict__ out, int n4) {
  int i = blockIdx.x * blockDim.x + threadIdx.x;
  if (i < n4) {
    float4 f = ((const float4*)in)[i];
    v4bf o;
    o[0] = (__bf16)f.x; o[1] = (__bf16)f.y; o[2] = (__bf16)f.z; o[3] = (__bf16)f.w;
    ((v4bf*)out)[i] = o;
  }
}

// ---------------- fused QKV projection + RoPE + V-transpose ----------------
// M = B*S = 4096, K = 2048, N = 3072 combined [Q:0..2047 | K:2048..2559 | V:2560..3071]
// WG: 256 threads (8 waves as 4Mx2N), macro-tile 256(M) x 128(N); wave tile 64x64.
// Tiles double-buffered in LDS via async-to-LDS copies.
__global__ void __launch_bounds__(256)
qkv_rope_kernel(const __bf16* __restrict__ xb,
                const __bf16* __restrict__ wqb,
                const __bf16* __restrict__ wkb,
                const __bf16* __restrict__ wvb,
                __bf16* __restrict__ qbuf,   // [B][NH][S][HD]
                __bf16* __restrict__ kbuf,   // [B][NKV][S][HD]
                __bf16* __restrict__ vtbuf)  // [B][NKV][HD][S]  (transposed)
{
  __shared__ __align__(32) __bf16 aTile[2][256 * LDS_ROW];  // 48 KB
  __shared__ __align__(32) __bf16 bTile[2][128 * LDS_ROW];  // 24 KB

  const int tid   = threadIdx.x;
  const int lane  = tid & 31;
  const int wave  = tid >> 5;
  const int l15   = lane & 15;
  const int h     = lane >> 4;
  const int waveM = wave & 3;
  const int waveN = wave >> 2;
  const int Mblk  = blockIdx.y * 256;
  const int Nblk  = blockIdx.x * 128;

  const __bf16* wptr;
  int nOff, region;
  if (Nblk < 2048)      { wptr = wqb; nOff = Nblk;        region = 0; }
  else if (Nblk < 2560) { wptr = wkb; nOff = Nblk - 2048; region = 1; }
  else                  { wptr = wvb; nOff = Nblk - 2560; region = 2; }

  // per-thread async copy assignments
  const __bf16* aSrc = xb + (size_t)(Mblk + tid) * HIDDEN;                       // 1 row/thread
  const __bf16* bSrc = wptr + (size_t)(nOff + (tid >> 1)) * HIDDEN + (tid & 1) * 16;  // half row
  unsigned aDst[2], bDst[2];
#pragma unroll
  for (int bfi = 0; bfi < 2; ++bfi) {
    aDst[bfi] = lds_off(&aTile[bfi][tid * LDS_ROW]);
    bDst[bfi] = lds_off(&bTile[bfi][(tid >> 1) * LDS_ROW + (tid & 1) * 16]);
  }

  v8f acc[4][4];
#pragma unroll
  for (int i = 0; i < 4; ++i)
#pragma unroll
    for (int j = 0; j < 4; ++j)
#pragma unroll
      for (int r = 0; r < 8; ++r) acc[i][j][r] = 0.0f;

  async_copy64B(aDst[0], aSrc);
  async_copy32B(bDst[0], bSrc);

  for (int kb = 0, it = 0; kb < HIDDEN; kb += 32, ++it) {
    const int cur = it & 1;
    if (kb + 32 < HIDDEN) {
      async_copy64B(aDst[cur ^ 1], aSrc + kb + 32);
      async_copy32B(bDst[cur ^ 1], bSrc + kb + 32);
      async_wait6();   // previous stage landed in LDS
    } else {
      async_wait0();
    }
    __syncthreads();   // all waves' async writes for `cur` visible

    const __bf16* aL = &aTile[cur][(waveM * 64) * LDS_ROW];
    const __bf16* bL = &bTile[cur][(waveN * 64) * LDS_ROW];
    v16bf a[4];
#pragma unroll
    for (int ms = 0; ms < 4; ++ms) {
      const __bf16* p = aL + (ms * 16 + l15) * LDS_ROW + 16 * h;
      a[ms] = load_a_frag(p, p + 8);
    }
#pragma unroll
    for (int ns = 0; ns < 4; ++ns) {
      v16bf bfr = load_b_frag(bL + (ns * 16 + l15) * LDS_ROW + 16 * h);
#pragma unroll
      for (int ms = 0; ms < 4; ++ms)
        acc[ms][ns] = WMMA_BF16(a[ms], bfr, acc[ms][ns]);
    }
    __syncthreads();   // done reading `cur`; next iter may overwrite it
  }

  const int b_ = Mblk >> 11;  // tiles never straddle batches (2048 % 256 == 0)
  const int Mw = Mblk + waveM * 64;
  const int Nw = Nblk + waveN * 64;
  if (region == 2) {
    // V: store transposed [b][kh][d][s]; 8 consecutive s per lane -> one 16B store
#pragma unroll
    for (int ns = 0; ns < 4; ++ns) {
      const int nn = Nw - 2560 + ns * 16 + l15;
      const int kh = nn >> 6;
      const int d  = nn & 63;
#pragma unroll
      for (int ms = 0; ms < 4; ++ms) {
        const int s0 = (Mw + ms * 16 + 8 * h) & (SEQ - 1);
        v8bf pk;
#pragma unroll
        for (int r = 0; r < 8; ++r) pk[r] = (__bf16)acc[ms][ns][r];
        *(v8bf*)(vtbuf + ((size_t)((b_ * NKV + kh) * HD + d)) * SEQ + s0) = pk;
      }
    }
  } else {
    // Q / K with RoPE: pair partner (d^1) is the adjacent lane
#pragma unroll
    for (int ns = 0; ns < 4; ++ns) {
      const int n = Nw + ns * 16 + l15;
      const int d = n & 63;
      const float freq = __expf(-(float)(d & 62) * (9.210340371976184f / 64.0f));
#pragma unroll
      for (int ms = 0; ms < 4; ++ms) {
#pragma unroll
        for (int r = 0; r < 8; ++r) {
          const int m = Mw + ms * 16 + r + 8 * h;
          const int s = m & (SEQ - 1);
          float v = acc[ms][ns][r];
          float o = __shfl_xor(v, 1, 32);
          float sn, cs;
          __sincosf((float)s * freq, &sn, &cs);
          float res = (d & 1) ? (o * sn + v * cs) : (v * cs - o * sn);
          if (region == 0) {
            qbuf[((size_t)((b_ * NH + (n >> 6)) * SEQ + s)) * HD + d] = (__bf16)res;
          } else {
            kbuf[((size_t)((b_ * NKV + ((n - 2048) >> 6)) * SEQ + s)) * HD + d] = (__bf16)res;
          }
        }
      }
    }
  }
}

// ---------------- causal GQA flash attention ----------------
// grid.x = B * NH * (S/64); 4 waves, each owns 16 q-rows of a 64-row q tile.
__global__ void __launch_bounds__(128)
gqa_attn_kernel(const __bf16* __restrict__ qbuf,
                const __bf16* __restrict__ kbuf,
                const __bf16* __restrict__ vtbuf,
                __bf16* __restrict__ attn)   // [B*S][NH*HD]
{
  __shared__ __align__(16) __bf16 plds[4][16][80];  // per-wave padded P tile

  const int lane = threadIdx.x & 31;
  const int wave = threadIdx.x >> 5;
  const int l15  = lane & 15;
  const int h4   = lane >> 4;
  const int wg   = blockIdx.x;
  const int qt   = wg & 31;
  const int head = (wg >> 5) & 31;
  const int b    = wg >> 10;
  const int kh   = head >> 2;  // G = 4
  const int qbase = qt * 64 + wave * 16;

  const __bf16* qrow = qbuf + ((size_t)((b * NH + head) * SEQ + qbase + l15)) * HD;
  const v16bf qf0 = load_a_frag(qrow + 16 * h4,      qrow + 8 + 16 * h4);
  const v16bf qf1 = load_a_frag(qrow + 32 + 16 * h4, qrow + 40 + 16 * h4);

  float mrow[8], lrow[8];
  v8f o[4];
#pragma unroll
  for (int r = 0; r < 8; ++r) { mrow[r] = -3.0e38f; lrow[r] = 0.0f; }
#pragma unroll
  for (int d = 0; d < 4; ++d)
#pragma unroll
    for (int r = 0; r < 8; ++r) o[d][r] = 0.0f;

  const __bf16* kbase = kbuf  + ((size_t)(b * NKV + kh)) * SEQ * HD;
  const __bf16* vbase = vtbuf + ((size_t)(b * NKV + kh)) * HD * SEQ;

  for (int tt = 0; tt <= qt; ++tt) {
    const int t0 = tt * 64;
    if (tt < qt) {  // prefetch next K/V tiles into cache
      __builtin_prefetch(kbase + (size_t)(t0 + 64 + lane) * HD, 0, 3);
      __builtin_prefetch(vbase + (size_t)lane * SEQ + t0 + 64, 0, 3);
    }
    v8f sc[4];
#pragma unroll
    for (int j = 0; j < 4; ++j) {
      const __bf16* kr = kbase + (size_t)(t0 + j * 16 + l15) * HD;
      v16bf kb0 = load_b_frag(kr + 16 * h4);
      v16bf kb1 = load_b_frag(kr + 32 + 16 * h4);
      v8f c;
#pragma unroll
      for (int r = 0; r < 8; ++r) c[r] = 0.0f;
      c = WMMA_BF16(qf0, kb0, c);
      c = WMMA_BF16(qf1, kb1, c);
      sc[j] = c;
    }
    // scale + causal mask (diagonal tile only)
#pragma unroll
    for (int j = 0; j < 4; ++j)
#pragma unroll
      for (int r = 0; r < 8; ++r) {
        float s_ = sc[j][r] * 0.125f;
        if (tt == qt && (t0 + j * 16 + l15) > (qbase + r + 8 * h4)) s_ = -3.0e38f;
        sc[j][r] = s_;
      }
    // online softmax: rows live per-lane-half -> reduce over 16 lanes
#pragma unroll
    for (int r = 0; r < 8; ++r) {
      float mx = fmaxf(fmaxf(sc[0][r], sc[1][r]), fmaxf(sc[2][r], sc[3][r]));
#pragma unroll
      for (int msk = 1; msk < 16; msk <<= 1) mx = fmaxf(mx, __shfl_xor(mx, msk, 32));
      const float mnew   = fmaxf(mrow[r], mx);
      const float sc_old = __expf(mrow[r] - mnew);
      float sum = 0.0f;
#pragma unroll
      for (int j = 0; j < 4; ++j) {
        float p = __expf(sc[j][r] - mnew);
        sc[j][r] = p;
        sum += p;
      }
#pragma unroll
      for (int msk = 1; msk < 16; msk <<= 1) sum += __shfl_xor(sum, msk, 32);
      lrow[r] = lrow[r] * sc_old + sum;
      mrow[r] = mnew;
#pragma unroll
      for (int d = 0; d < 4; ++d) o[d][r] *= sc_old;
    }
    // transpose P (C layout -> A layout) through LDS
#pragma unroll
    for (int j = 0; j < 4; ++j)
#pragma unroll
      for (int r = 0; r < 8; ++r)
        plds[wave][r + 8 * h4][j * 16 + l15] = (__bf16)sc[j][r];
    __syncthreads();
    const __bf16* pr = &plds[wave][l15][0];
    v16bf pf0 = load_a_frag(pr + 16 * h4,      pr + 8 + 16 * h4);
    v16bf pf1 = load_a_frag(pr + 32 + 16 * h4, pr + 40 + 16 * h4);
    __syncthreads();
    // O += P @ V   (V pre-transposed: lane d gets t-contiguous 16 bf16)
#pragma unroll
    for (int d = 0; d < 4; ++d) {
      const __bf16* vr = vbase + (size_t)(d * 16 + l15) * SEQ + t0;
      v16bf vb0 = load_b_frag(vr + 16 * h4);
      v16bf vb1 = load_b_frag(vr + 32 + 16 * h4);
      o[d] = WMMA_BF16(pf0, vb0, o[d]);
      o[d] = WMMA_BF16(pf1, vb1, o[d]);
    }
  }
  // normalize + store attn output [b*S + s][head*64 + d]
#pragma unroll
  for (int d = 0; d < 4; ++d)
#pragma unroll
    for (int r = 0; r < 8; ++r) {
      const int s = qbase + r + 8 * h4;
      attn[((size_t)(b * SEQ + s)) * (NH * HD) + head * HD + d * 16 + l15] =
          (__bf16)(o[d][r] / lrow[r]);
    }
}

// ---------------- output projection: out = attn @ wo^T (fp32 out) ----------------
// Same async-staged 256x128 macro-tile GEMM; fp32 stores to d_out.
__global__ void __launch_bounds__(256)
out_proj_kernel(const __bf16* __restrict__ attnb,
                const __bf16* __restrict__ wob,
                float* __restrict__ out) {
  __shared__ __align__(32) __bf16 aTile[2][256 * LDS_ROW];
  __shared__ __align__(32) __bf16 bTile[2][128 * LDS_ROW];

  const int tid   = threadIdx.x;
  const int lane  = tid & 31;
  const int wave  = tid >> 5;
  const int l15   = lane & 15;
  const int h     = lane >> 4;
  const int waveM = wave & 3;
  const int waveN = wave >> 2;
  const int Mblk  = blockIdx.y * 256;
  const int Nblk  = blockIdx.x * 128;

  const __bf16* aSrc = attnb + (size_t)(Mblk + tid) * HIDDEN;
  const __bf16* bSrc = wob + (size_t)(Nblk + (tid >> 1)) * HIDDEN + (tid & 1) * 16;
  unsigned aDst[2], bDst[2];
#pragma unroll
  for (int bfi = 0; bfi < 2; ++bfi) {
    aDst[bfi] = lds_off(&aTile[bfi][tid * LDS_ROW]);
    bDst[bfi] = lds_off(&bTile[bfi][(tid >> 1) * LDS_ROW + (tid & 1) * 16]);
  }

  v8f acc[4][4];
#pragma unroll
  for (int i = 0; i < 4; ++i)
#pragma unroll
    for (int j = 0; j < 4; ++j)
#pragma unroll
      for (int r = 0; r < 8; ++r) acc[i][j][r] = 0.0f;

  async_copy64B(aDst[0], aSrc);
  async_copy32B(bDst[0], bSrc);

  for (int kb = 0, it = 0; kb < HIDDEN; kb += 32, ++it) {
    const int cur = it & 1;
    if (kb + 32 < HIDDEN) {
      async_copy64B(aDst[cur ^ 1], aSrc + kb + 32);
      async_copy32B(bDst[cur ^ 1], bSrc + kb + 32);
      async_wait6();
    } else {
      async_wait0();
    }
    __syncthreads();

    const __bf16* aL = &aTile[cur][(waveM * 64) * LDS_ROW];
    const __bf16* bL = &bTile[cur][(waveN * 64) * LDS_ROW];
    v16bf a[4];
#pragma unroll
    for (int ms = 0; ms < 4; ++ms) {
      const __bf16* p = aL + (ms * 16 + l15) * LDS_ROW + 16 * h;
      a[ms] = load_a_frag(p, p + 8);
    }
#pragma unroll
    for (int ns = 0; ns < 4; ++ns) {
      v16bf bfr = load_b_frag(bL + (ns * 16 + l15) * LDS_ROW + 16 * h);
#pragma unroll
      for (int ms = 0; ms < 4; ++ms)
        acc[ms][ns] = WMMA_BF16(a[ms], bfr, acc[ms][ns]);
    }
    __syncthreads();
  }

#pragma unroll
  for (int ms = 0; ms < 4; ++ms)
#pragma unroll
    for (int ns = 0; ns < 4; ++ns)
#pragma unroll
      for (int r = 0; r < 8; ++r)
        out[(size_t)(Mblk + waveM * 64 + ms * 16 + r + 8 * h) * HIDDEN +
            Nblk + waveN * 64 + ns * 16 + l15] = acc[ms][ns][r];
}

// ---------------- host side ----------------
extern "C" void kernel_launch(void* const* d_in, const int* in_sizes, int n_in,
                              void* d_out, int out_size, void* d_ws, size_t ws_size,
                              hipStream_t stream) {
  (void)in_sizes; (void)n_in; (void)out_size; (void)ws_size;
  const float* x  = (const float*)d_in[0];
  const float* wq = (const float*)d_in[1];
  const float* wk = (const float*)d_in[2];
  const float* wv = (const float*)d_in[3];
  const float* wo = (const float*)d_in[4];
  float* out = (float*)d_out;

  char* ws = (char*)d_ws;
  __bf16* xb    = (__bf16*)(ws + 0);          // 16 MB
  __bf16* wqb   = (__bf16*)(ws + 16777216);   //  8 MB
  __bf16* wkb   = (__bf16*)(ws + 25165824);   //  2 MB
  __bf16* wvb   = (__bf16*)(ws + 27262976);   //  2 MB
  __bf16* wob   = (__bf16*)(ws + 29360128);   //  8 MB
  __bf16* qbuf  = (__bf16*)(ws + 37748736);   // 16 MB
  __bf16* kbuf  = (__bf16*)(ws + 54525952);   //  4 MB
  __bf16* vtbuf = (__bf16*)(ws + 58720256);   //  4 MB
  __bf16* attnb = (__bf16*)(ws + 62914560);   // 16 MB

  const int nx = BATCH * SEQ * HIDDEN / 4;    // float4 groups
  const int nq = NH * HD * HIDDEN / 4;
  const int nk = NKV * HD * HIDDEN / 4;
  cvt_bf16_kernel<<<(nx + 255) / 256, 256, 0, stream>>>(x,  xb,  nx);
  cvt_bf16_kernel<<<(nq + 255) / 256, 256, 0, stream>>>(wq, wqb, nq);
  cvt_bf16_kernel<<<(nk + 255) / 256, 256, 0, stream>>>(wk, wkb, nk);
  cvt_bf16_kernel<<<(nk + 255) / 256, 256, 0, stream>>>(wv, wvb, nk);
  cvt_bf16_kernel<<<(nq + 255) / 256, 256, 0, stream>>>(wo, wob, nq);

  qkv_rope_kernel<<<dim3(3072 / 128, 4096 / 256), 256, 0, stream>>>(
      xb, wqb, wkb, wvb, qbuf, kbuf, vtbuf);
  gqa_attn_kernel<<<dim3(BATCH * NH * (SEQ / 64)), 128, 0, stream>>>(
      qbuf, kbuf, vtbuf, attnb);
  out_proj_kernel<<<dim3(HIDDEN / 128, (BATCH * SEQ) / 256), 256, 0, stream>>>(
      attnb, wob, out);
}
